// Parallel_79697413144642
// MI455X (gfx1250) — matrix operand order repaired and checked by reading, not compiled, not run on hardware
//
#include <hip/hip_runtime.h>
#include <hip/hip_bf16.h>

#define DEVFN __device__ __forceinline__

typedef __bf16 bf16_t;
typedef bf16_t bf16x16 __attribute__((ext_vector_type(16)));
typedef float  f32x8   __attribute__((ext_vector_type(8)));

namespace {
constexpr int   N_NODES  = 100000;
constexpr int   N_EDGES  = 1600000;
constexpr int   F_IN     = 64;
constexpr int   R_REL    = 8;
constexpr int   G_GRAPHS = 256;
constexpr int   H_HEADS  = 4;
constexpr int   C_CH     = 32;
constexpr int   HC       = 128;           // H*C
constexpr int   NT_M     = N_NODES / 16;  // 6250 exact
constexpr int   CAT1_COLS  = HC + R_REL * C_CH + C_CH; // 416 = gat_w | rw1[0..7] | root1
constexpr int   CAT1_TILES = CAT1_COLS / 16;           // 26
constexpr int   CAT1_PAIRS = CAT1_TILES / 2;           // 13 (2 col tiles per wave)
constexpr int   CAT2_COLS  = R_REL * 16 + 16;          // 144 = rw2[0..7] | root2
constexpr int   CAT2_TILES = CAT2_COLS / 16;           // 9
constexpr float NEG_ATT  = 0.2f;
constexpr float NEG_SL   = 0.01f;
constexpr unsigned ENC_NEG_INF = 0x007FFFFFu; // enc(-inf)
}

DEVFN float lrelu(float v, float s) { return v > 0.0f ? v : v * s; }

DEVFN bf16_t f2bf(float f) { // fp32 -> bf16, round-to-nearest-even
  unsigned u = __float_as_uint(f);
  u += 0x7FFFu + ((u >> 16) & 1u);
  unsigned short hs = (unsigned short)(u >> 16);
  bf16_t r;
  __builtin_memcpy(&r, &hs, 2);
  return r;
}

// order-preserving float <-> uint encoding for atomicMax-based segment max
DEVFN unsigned enc_f32(float f) {
  unsigned u = __float_as_uint(f);
  return (u & 0x80000000u) ? ~u : (u | 0x80000000u);
}
DEVFN float dec_f32(unsigned e) {
  unsigned u = (e & 0x80000000u) ? (e & 0x7FFFFFFFu) : ~e;
  return __uint_as_float(u);
}

// ---------------- init ----------------
__global__ void k_fill_u32(unsigned* __restrict__ p, long n, unsigned v) {
  long i = (long)blockIdx.x * blockDim.x + threadIdx.x;
  if (i < n) p[i] = v;
}

// ---------------- pack x (fp32 -> bf16 once; removes per-tile cvt VALU from GEMM1) ----------
__global__ void k_pack_x(const float* __restrict__ x, bf16_t* __restrict__ xb, long n) {
  long i = (long)blockIdx.x * blockDim.x + threadIdx.x;
  if (i < n) xb[i] = f2bf(x[i]);
}

// ---------------- weight packing: column-major bf16 for contiguous B loads ----------------
__global__ void k_pack_weights(const float* __restrict__ gat_w, const float* __restrict__ rw1,
                               const float* __restrict__ root1, const float* __restrict__ dense1_w,
                               const float* __restrict__ rw2, const float* __restrict__ root2,
                               bf16_t* __restrict__ Bp1, bf16_t* __restrict__ Bpd1,
                               bf16_t* __restrict__ Bp2) {
  int i = blockIdx.x * blockDim.x + threadIdx.x;
  const int n1 = CAT1_COLS * 64, n2 = 16 * 128, n3 = CAT2_COLS * 32;
  if (i < n1) {
    int col = i / 64, k = i % 64;
    float v;
    if (col < 128)      v = gat_w[k * 128 + col];
    else if (col < 384) { int r = (col - 128) >> 5, c = (col - 128) & 31;
                          v = rw1[((size_t)r * 64 + k) * 32 + c]; }
    else                v = root1[k * 32 + (col - 384)];
    Bp1[(size_t)col * 64 + k] = f2bf(v);
  } else if (i < n1 + n2) {
    int j = i - n1, col = j / 128, k = j % 128;
    Bpd1[(size_t)col * 128 + k] = f2bf(dense1_w[k * 16 + col]);
  } else if (i < n1 + n2 + n3) {
    int j = i - n1 - n2, col = j / 32, k = j % 32;
    float v;
    if (col < 128) { int r = col >> 4, c = col & 15; v = rw2[((size_t)r * 32 + k) * 16 + c]; }
    else           v = root2[k * 16 + (col - 128)];
    Bp2[(size_t)col * 32 + k] = f2bf(v);
  }
}

// hoisted (wave-uniform) epilogue for the fused cat-1 GEMM
DEVFN void store_cat1(int ct, int tileM, int lane, const f32x8& acc, const float* rb1,
                      float* __restrict__ h, float* __restrict__ xw1,
                      float* __restrict__ z1root) {
  const int mn = lane & 15;
  const int rowoff = (lane >> 4) * 8;
  float* base;
  size_t stride;
  float bias = 0.0f;
  if (ct < 8) {
    base = h + (size_t)(tileM * 16 + rowoff) * HC + ct * 16 + mn;
    stride = HC;
  } else if (ct < 24) {
    int rel = (ct - 8) >> 1;
    int c32 = ((ct - 8) & 1) * 16 + mn;
    base = xw1 + ((size_t)rel * N_NODES + tileM * 16 + rowoff) * C_CH + c32;
    stride = C_CH;
  } else {
    int c32 = (ct - 24) * 16 + mn;
    base = z1root + (size_t)(tileM * 16 + rowoff) * C_CH + c32;
    stride = C_CH;
    bias = rb1[c32];
  }
#pragma unroll
  for (int j = 0; j < 8; ++j) base[(size_t)j * stride] = acc[j] + bias;
}

// ---------------- fused GEMM 1: [h | xw1(r=0..7) | z1root+rb1] = x @ Bp1 ----------------
// one wave = 16 rows x 2 column tiles; A (bf16-prepacked) registers feed 4 WMMAs
__global__ __launch_bounds__(32)
void k_gemm1(const bf16_t* __restrict__ xb, const bf16_t* __restrict__ Bp1,
             const float* __restrict__ rb1, float* __restrict__ h,
             float* __restrict__ xw1, float* __restrict__ z1root) {
  const int lane  = threadIdx.x & 31;
  const int tileM = blockIdx.x;
  const int ct0   = blockIdx.y * 2;
  const int mn    = lane & 15;            // A row within tile / B column within tile
  const int khA   = (lane >> 4) * 16;     // A: lanes 0-15 K=0..15, lanes 16-31 K=16..31
  const int ksB   = (lane >> 4) * 16;     // B: lanes 0-15 K=0..15, lanes 16-31 K=16..31
  const bf16_t* arow  = xb  + (size_t)(tileM * 16 + mn) * F_IN;
  const bf16_t* bcol0 = Bp1 + (size_t)(ct0 * 16 + mn) * F_IN;
  const bf16_t* bcol1 = bcol0 + (size_t)16 * F_IN;
  f32x8 acc0 = {}, acc1 = {};
#pragma unroll
  for (int ks = 0; ks < F_IN; ks += 32) {
    bf16x16 a, b0, b1;
#pragma unroll
    for (int i = 0; i < 16; ++i) {
      a[i]     = arow[ks + khA + i];
    }
#pragma unroll
    for (int i = 0; i < 16; ++i) {
      b0[i] = bcol0[ks + ksB + i];
      b1[i] = bcol1[ks + ksB + i];
    }
    acc0 = __builtin_amdgcn_wmma_f32_16x16x32_bf16(false, a, false, b0, (short)0, acc0, false, false);
    acc1 = __builtin_amdgcn_wmma_f32_16x16x32_bf16(false, a, false, b1, (short)0, acc1, false, false);
  }
  store_cat1(ct0,     tileM, lane, acc0, rb1, h, xw1, z1root);
  store_cat1(ct0 + 1, tileM, lane, acc1, rb1, h, xw1, z1root);
}

// ---------------- GAT attention scalar scores ----------------
__global__ void k_att(const float* __restrict__ h, const float* __restrict__ att_src,
                      const float* __restrict__ att_dst, float* __restrict__ a_src,
                      float* __restrict__ a_dst) {
  int i = blockIdx.x * blockDim.x + threadIdx.x;
  if (i >= N_NODES * H_HEADS) return;
  int n = i >> 2, hd = i & 3;
  const float* hp = h + (size_t)n * HC + hd * C_CH;
  float ss = 0.0f, sd = 0.0f;
#pragma unroll 8
  for (int c = 0; c < C_CH; ++c) {
    float v = hp[c];
    ss += v * att_src[hd * C_CH + c];
    sd += v * att_dst[hd * C_CH + c];
  }
  a_src[i] = ss;
  a_dst[i] = sd;
}

// ---------------- segment softmax (max pass) ----------------
__global__ void k_edge_max(const int* __restrict__ src, const int* __restrict__ dst,
                           const float* __restrict__ a_src, const float* __restrict__ a_dst,
                           float* __restrict__ e_buf, unsigned* __restrict__ m_enc) {
  int e = blockIdx.x * blockDim.x + threadIdx.x;
  if (e >= N_EDGES) return;
  int s = src[e], d = dst[e];
#pragma unroll
  for (int hd = 0; hd < H_HEADS; ++hd) {
    float v = lrelu(a_src[s * 4 + hd] + a_dst[d * 4 + hd], NEG_ATT);
    e_buf[(size_t)e * 4 + hd] = v;
    atomicMax(&m_enc[d * 4 + hd], enc_f32(v));
  }
}

// ---------------- segment softmax (exp + sum pass) ----------------
__global__ void k_edge_expsum(const int* __restrict__ dst, float* __restrict__ e_buf,
                              const unsigned* __restrict__ m_enc, float* __restrict__ ssum) {
  int e = blockIdx.x * blockDim.x + threadIdx.x;
  if (e >= N_EDGES) return;
  int d = dst[e];
#pragma unroll
  for (int hd = 0; hd < H_HEADS; ++hd) {
    float p = __expf(e_buf[(size_t)e * 4 + hd] - dec_f32(m_enc[d * 4 + hd]));
    e_buf[(size_t)e * 4 + hd] = p;
    atomicAdd(&ssum[d * 4 + hd], p);
  }
}

// ---------------- GAT message aggregation (thread per edge-quad, float4) ----------------
__global__ void k_gat_agg4(const int* __restrict__ src, const int* __restrict__ dst,
                           const float* __restrict__ e_buf, const float* __restrict__ ssum,
                           const float* __restrict__ h, float* __restrict__ gat_out) {
  long idx = (long)blockIdx.x * blockDim.x + threadIdx.x;
  if (idx >= (long)N_EDGES * (HC / 4)) return;
  int e = (int)(idx >> 5), q = (int)(idx & 31), hd = q >> 3;
  int s = src[e], d = dst[e];
  float alpha = e_buf[(size_t)e * 4 + hd] / ssum[d * 4 + hd];
  const float4 hv = *(const float4*)(h + (size_t)s * HC + q * 4);
  float* gp = gat_out + (size_t)d * HC + q * 4;
  atomicAdd(gp + 0, hv.x * alpha);
  atomicAdd(gp + 1, hv.y * alpha);
  atomicAdd(gp + 2, hv.z * alpha);
  atomicAdd(gp + 3, hv.w * alpha);
}

// ---------------- dense1: h2 = lrelu(lrelu(gat_out+bias) @ dense1_w + b) ----------------
__global__ __launch_bounds__(32)
void k_dense1(const float* __restrict__ gat_out, const float* __restrict__ gat_bias,
              const bf16_t* __restrict__ Bpd1, const float* __restrict__ d1b,
              float* __restrict__ h2) {
  const int lane  = threadIdx.x & 31;
  const int tileM = blockIdx.x;
  const int mn    = lane & 15;
  const int khA   = (lane >> 4) * 16;
  const int ksB   = (lane >> 4) * 16;
  const int row   = tileM * 16 + mn;
  const float*  arow = gat_out + (size_t)row * HC;
  const bf16_t* bcol = Bpd1 + (size_t)mn * HC;
  f32x8 acc = {};
#pragma unroll
  for (int ks = 0; ks < HC; ks += 32) {
    bf16x16 a, b;
#pragma unroll
    for (int i = 0; i < 16; ++i) {
      int k0 = ks + khA + i;
      a[i]     = f2bf(lrelu(arow[k0] + gat_bias[k0], NEG_SL));
    }
#pragma unroll
    for (int i = 0; i < 16; ++i) b[i] = bcol[ks + ksB + i];
    acc = __builtin_amdgcn_wmma_f32_16x16x32_bf16(false, a, false, b, (short)0, acc, false, false);
  }
#pragma unroll
  for (int j = 0; j < 8; ++j) {
    int r = tileM * 16 + j + ((lane >> 4) * 8);
    h2[(size_t)r * 16 + mn] = lrelu(acc[j] + d1b[mn], NEG_SL);
  }
}

// ---------------- graph segment max of h2 ----------------
__global__ void k_gmax(const float* __restrict__ h2, const int* __restrict__ batch,
                       unsigned* __restrict__ gmax_enc) {
  long idx = (long)blockIdx.x * blockDim.x + threadIdx.x;
  if (idx >= (long)N_NODES * 16) return;
  int n = (int)(idx >> 4), c = (int)(idx & 15);
  atomicMax(&gmax_enc[batch[n] * 16 + c], enc_f32(h2[idx]));
}

// ---------------- RGCN (dst, relation) counts ----------------
__global__ void k_cnt(const int* __restrict__ dst, const int* __restrict__ et,
                      float* __restrict__ cnt) {
  int e = blockIdx.x * blockDim.x + threadIdx.x;
  if (e >= N_EDGES) return;
  atomicAdd(&cnt[(size_t)dst[e] * R_REL + et[e]], 1.0f);
}

// ---------------- RGCN edge aggregation, float4 per thread ----------------
__global__ void k_rgcn_agg4(const int* __restrict__ src, const int* __restrict__ dst,
                            const int* __restrict__ et, const float* __restrict__ cnt,
                            const float* __restrict__ xw, float* __restrict__ agg,
                            int logq, int cols) {
  long idx = (long)blockIdx.x * blockDim.x + threadIdx.x;
  if (idx >= (long)N_EDGES << logq) return;
  int e = (int)(idx >> logq), q = (int)(idx & ((1 << logq) - 1));
  int s = src[e], d = dst[e], t = et[e];
  float inv = 1.0f / cnt[(size_t)d * R_REL + t];
  const float4 v = *(const float4*)(xw + ((size_t)t * N_NODES + s) * cols + q * 4);
  float* ap = agg + (size_t)d * cols + q * 4;
  atomicAdd(ap + 0, v.x * inv);
  atomicAdd(ap + 1, v.y * inv);
  atomicAdd(ap + 2, v.z * inv);
  atomicAdd(ap + 3, v.w * inv);
}

// ---------------- z1 = relu(agg + root_part) -> bf16 (feeds GEMM2 A operand) ------------
__global__ void k_add_relu_bf16(const float* __restrict__ agg, const float* __restrict__ rootp,
                                bf16_t* __restrict__ zb, long n) {
  long i = (long)blockIdx.x * blockDim.x + threadIdx.x;
  if (i >= n) return;
  float v = agg[i] + rootp[i];
  zb[i] = f2bf(v > 0.0f ? v : 0.0f);
}

// ---------------- z2 = relu(agg + root_part) (f32, in place) ----------------
__global__ void k_add_relu(float* __restrict__ agg, const float* __restrict__ rootp, long n) {
  long i = (long)blockIdx.x * blockDim.x + threadIdx.x;
  if (i >= n) return;
  float v = agg[i] + rootp[i];
  agg[i] = v > 0.0f ? v : 0.0f;
}

// ---------------- fused GEMM 2: [xw2(r=0..7) | z2root+rb2] = z1 @ Bp2 ----------------
__global__ __launch_bounds__(32)
void k_gemm2(const bf16_t* __restrict__ z1b, const bf16_t* __restrict__ Bp2,
             const float* __restrict__ rb2, float* __restrict__ xw2,
             float* __restrict__ z2root) {
  const int lane  = threadIdx.x & 31;
  const int tileM = blockIdx.x;
  const int ct    = blockIdx.y;
  const int mn    = lane & 15;
  const int khA   = (lane >> 4) * 16;
  const int ksB   = (lane >> 4) * 16;
  const bf16_t* arow = z1b + (size_t)(tileM * 16 + mn) * C_CH;
  const bf16_t* bcol = Bp2 + (size_t)(ct * 16 + mn) * C_CH;
  bf16x16 a, b;
#pragma unroll
  for (int i = 0; i < 16; ++i) {
    a[i]     = arow[khA + i];
  }
#pragma unroll
  for (int i = 0; i < 16; ++i) b[i] = bcol[ksB + i];
  f32x8 acc = {};
  acc = __builtin_amdgcn_wmma_f32_16x16x32_bf16(false, a, false, b, (short)0, acc, false, false);
  // hoisted, wave-uniform epilogue
  const int rowoff = (lane >> 4) * 8;
  float* base;
  float bias = 0.0f;
  if (ct < 8) {
    base = xw2 + ((size_t)ct * N_NODES + tileM * 16 + rowoff) * 16 + mn;
  } else {
    base = z2root + (size_t)(tileM * 16 + rowoff) * 16 + mn;
    bias = rb2[mn];
  }
#pragma unroll
  for (int j = 0; j < 8; ++j) base[(size_t)j * 16] = acc[j] + bias;
}

// ---------------- graph segment sum + node counts ----------------
__global__ void k_gsum(const float* __restrict__ z2, const int* __restrict__ batch,
                       float* __restrict__ gsum, float* __restrict__ ncnt) {
  long idx = (long)blockIdx.x * blockDim.x + threadIdx.x;
  if (idx >= (long)N_NODES * 16) return;
  int n = (int)(idx >> 4), c = (int)(idx & 15);
  int g = batch[n];
  atomicAdd(&gsum[g * 16 + c], z2[idx]);
  if (c == 0) atomicAdd(&ncnt[g], 1.0f);
}

// ---------------- final fuse: [gmax | gmean] @ dense_w + dense_b ----------------
__global__ void k_fuse(const unsigned* __restrict__ gmax_enc, const float* __restrict__ gsum,
                       const float* __restrict__ ncnt, const float* __restrict__ dense_w,
                       const float* __restrict__ dense_b, float* __restrict__ out) {
  int g = blockIdx.x * blockDim.x + threadIdx.x;
  if (g >= G_GRAPHS) return;
  float acc = dense_b[0];
  float inv = 1.0f / ncnt[g];
#pragma unroll
  for (int j = 0; j < 16; ++j) acc += dec_f32(gmax_enc[g * 16 + j]) * dense_w[j];
#pragma unroll
  for (int j = 0; j < 16; ++j) acc += gsum[g * 16 + j] * inv * dense_w[16 + j];
  out[g] = acc;
}

extern "C" void kernel_launch(void* const* d_in, const int* in_sizes, int n_in,
                              void* d_out, int out_size, void* d_ws, size_t ws_size,
                              hipStream_t stream) {
  (void)in_sizes; (void)n_in; (void)out_size; (void)ws_size;
  const float* x         = (const float*)d_in[0];
  const int*   ei        = (const int*)d_in[1];
  const int*   etype     = (const int*)d_in[2];
  const int*   batch     = (const int*)d_in[3];
  const float* gat_w     = (const float*)d_in[4];
  const float* att_src_w = (const float*)d_in[5];
  const float* att_dst_w = (const float*)d_in[6];
  const float* gat_bias  = (const float*)d_in[7];
  const float* dense1_w  = (const float*)d_in[8];
  const float* dense1_b  = (const float*)d_in[9];
  const float* rw1       = (const float*)d_in[10];
  const float* root1     = (const float*)d_in[11];
  const float* rb1       = (const float*)d_in[12];
  const float* rw2       = (const float*)d_in[13];
  const float* root2     = (const float*)d_in[14];
  const float* rb2       = (const float*)d_in[15];
  const float* dense_w   = (const float*)d_in[16];
  const float* dense_b   = (const float*)d_in[17];
  const int* src = ei;
  const int* dst = ei + N_EDGES;
  float* out = (float*)d_out;

  char* wsp = (char*)d_ws;
  auto carve = [&](size_t bytes) { char* p = wsp; wsp += (bytes + 255) & ~(size_t)255; return p; };
  float*    h        = (float*)carve((size_t)N_NODES * HC * 4);
  float*    gat_out  = (float*)carve((size_t)N_NODES * HC * 4);
  float*    a_srcb   = (float*)carve((size_t)N_NODES * H_HEADS * 4);
  float*    a_dstb   = (float*)carve((size_t)N_NODES * H_HEADS * 4);
  unsigned* m_enc    = (unsigned*)carve((size_t)N_NODES * H_HEADS * 4);
  float*    ssum     = (float*)carve((size_t)N_NODES * H_HEADS * 4);
  float*    e_buf    = (float*)carve((size_t)N_EDGES * H_HEADS * 4);
  float*    h2       = (float*)carve((size_t)N_NODES * 16 * 4);
  float*    xw1      = (float*)carve((size_t)R_REL * N_NODES * C_CH * 4);
  float*    z1root   = (float*)carve((size_t)N_NODES * C_CH * 4);
  float*    agg1     = (float*)carve((size_t)N_NODES * C_CH * 4);
  float*    cnt      = (float*)carve((size_t)N_NODES * R_REL * 4);
  float*    xw2      = (float*)carve((size_t)R_REL * N_NODES * 16 * 4);
  float*    z2root   = (float*)carve((size_t)N_NODES * 16 * 4);
  float*    agg2     = (float*)carve((size_t)N_NODES * 16 * 4);     // becomes z2
  unsigned* gmax_enc = (unsigned*)carve((size_t)G_GRAPHS * 16 * 4);
  float*    gsum     = (float*)carve((size_t)G_GRAPHS * 16 * 4);
  float*    ncnt     = (float*)carve((size_t)G_GRAPHS * 4);
  bf16_t*   xb       = (bf16_t*)carve((size_t)N_NODES * F_IN * 2);
  bf16_t*   z1b      = (bf16_t*)carve((size_t)N_NODES * C_CH * 2);
  bf16_t*   Bp1      = (bf16_t*)carve((size_t)CAT1_COLS * 64 * 2);
  bf16_t*   Bpd1     = (bf16_t*)carve((size_t)16 * 128 * 2);
  bf16_t*   Bp2      = (bf16_t*)carve((size_t)CAT2_COLS * 32 * 2);

  auto nb = [](long n, int b) { return (unsigned)((n + b - 1) / b); };

  // per-call re-init of accumulators (graph-replay safe)
  k_fill_u32<<<nb((long)N_NODES * 4, 256), 256, 0, stream>>>(m_enc, (long)N_NODES * 4, ENC_NEG_INF);
  k_fill_u32<<<nb((long)N_NODES * 4, 256), 256, 0, stream>>>((unsigned*)ssum, (long)N_NODES * 4, 0u);
  k_fill_u32<<<nb((long)N_NODES * HC, 256), 256, 0, stream>>>((unsigned*)gat_out, (long)N_NODES * HC, 0u);
  k_fill_u32<<<nb((long)N_NODES * R_REL, 256), 256, 0, stream>>>((unsigned*)cnt, (long)N_NODES * R_REL, 0u);
  k_fill_u32<<<nb((long)N_NODES * C_CH, 256), 256, 0, stream>>>((unsigned*)agg1, (long)N_NODES * C_CH, 0u);
  k_fill_u32<<<nb((long)N_NODES * 16, 256), 256, 0, stream>>>((unsigned*)agg2, (long)N_NODES * 16, 0u);
  k_fill_u32<<<nb((long)G_GRAPHS * 16, 256), 256, 0, stream>>>(gmax_enc, (long)G_GRAPHS * 16, ENC_NEG_INF);
  k_fill_u32<<<nb((long)G_GRAPHS * 16, 256), 256, 0, stream>>>((unsigned*)gsum, (long)G_GRAPHS * 16, 0u);
  k_fill_u32<<<nb((long)G_GRAPHS, 256), 256, 0, stream>>>((unsigned*)ncnt, (long)G_GRAPHS, 0u);

  // operand packing (bf16)
  k_pack_x<<<nb((long)N_NODES * F_IN, 256), 256, 0, stream>>>(x, xb, (long)N_NODES * F_IN);
  k_pack_weights<<<nb(CAT1_COLS * 64 + 16 * 128 + CAT2_COLS * 32, 256), 256, 0, stream>>>(
      gat_w, rw1, root1, dense1_w, rw2, root2, Bp1, Bpd1, Bp2);

  // fused node-transform GEMM 1 (WMMA bf16, 2 col tiles / wave)
  k_gemm1<<<dim3(NT_M, CAT1_PAIRS), 32, 0, stream>>>(xb, Bp1, rb1, h, xw1, z1root);

  // GAT branch
  k_att<<<nb((long)N_NODES * H_HEADS, 256), 256, 0, stream>>>(h, att_src_w, att_dst_w, a_srcb, a_dstb);
  k_edge_max<<<nb(N_EDGES, 256), 256, 0, stream>>>(src, dst, a_srcb, a_dstb, e_buf, m_enc);
  k_edge_expsum<<<nb(N_EDGES, 256), 256, 0, stream>>>(dst, e_buf, m_enc, ssum);
  k_gat_agg4<<<nb((long)N_EDGES * (HC / 4), 256), 256, 0, stream>>>(src, dst, e_buf, ssum, h, gat_out);
  k_dense1<<<NT_M, 32, 0, stream>>>(gat_out, gat_bias, Bpd1, dense1_b, h2);
  k_gmax<<<nb((long)N_NODES * 16, 256), 256, 0, stream>>>(h2, batch, gmax_enc);

  // RGCN branch (layer 1)
  k_cnt<<<nb(N_EDGES, 256), 256, 0, stream>>>(dst, etype, cnt);
  k_rgcn_agg4<<<nb((long)N_EDGES * 8, 256), 256, 0, stream>>>(src, dst, etype, cnt, xw1, agg1, 3, C_CH);
  k_add_relu_bf16<<<nb((long)N_NODES * C_CH, 256), 256, 0, stream>>>(agg1, z1root, z1b,
                                                                     (long)N_NODES * C_CH);
  // layer 2
  k_gemm2<<<dim3(NT_M, CAT2_TILES), 32, 0, stream>>>(z1b, Bp2, rb2, xw2, z2root);
  k_rgcn_agg4<<<nb((long)N_EDGES * 4, 256), 256, 0, stream>>>(src, dst, etype, cnt, xw2, agg2, 2, 16);
  k_add_relu<<<nb((long)N_NODES * 16, 256), 256, 0, stream>>>(agg2, z2root, (long)N_NODES * 16);

  // graph mean + fuse
  k_gsum<<<nb((long)N_NODES * 16, 256), 256, 0, stream>>>(agg2, batch, gsum, ncnt);
  k_fuse<<<1, 256, 0, stream>>>(gmax_enc, gsum, ncnt, dense_w, dense_b, out);
}
